// GraphMatch_85048942396250
// MI455X (gfx1250) — compile-verified, matrix operand order checked by reading, not run on hardware
//
#include <hip/hip_runtime.h>
#include <math.h>

#define BSz 2
#define Nn  256
#define Dd  128
#define Kk  3

typedef __attribute__((ext_vector_type(16))) _Float16 v16h;
typedef __attribute__((ext_vector_type(8)))  _Float16 v8h;
typedef __attribute__((ext_vector_type(8)))  float    v8f;

// workspace layout (float units)
#define WS_CONST 0                       // 6*128 folded constants
#define WS_U     768                     // [BS*2][N][D]
#define WS_V     (WS_U   + BSz*2*Nn*Dd)
#define WS_ABS   (WS_V   + BSz*2*Nn*Dd)  // absorbed feats [BS*2][N][D]
#define WS_NBR   (WS_ABS + BSz*2*Nn*Dd)  // neighbor feats [BS*2][N][K][D]

// ---------------------------------------------------------------------------
// Fold BN + bias into per-channel (mul, add, w1) for both MLPs.
//   h = x@W0 + b0 ; bn = (h-m)*g/sqrt(v+eps)+be  =>  (x@W0)*A + ((b0-m)*A+be)
// ---------------------------------------------------------------------------
__global__ void fold_kernel(const float* sg, const float* sbe, const float* sm,
                            const float* sv, const float* sb0, const float* sw1,
                            const float* cg, const float* cbe, const float* cm,
                            const float* cv, const float* cb0, const float* cw1,
                            float* ws) {
  int c = threadIdx.x;
  if (c < Dd) {
    float A = sg[c] * rsqrtf(sv[c] + 1e-5f);
    ws[c]       = A;
    ws[128 + c] = (sb0[c] - sm[c]) * A + sbe[c];
    ws[256 + c] = sw1[c];
    float Ac = cg[c] * rsqrtf(cv[c] + 1e-5f);
    ws[384 + c] = Ac;
    ws[512 + c] = (cb0[c] - cm[c]) * Ac + cbe[c];
    ws[640 + c] = cw1[c];
  }
}

// ---------------------------------------------------------------------------
// Separable sim first layer: u = f @ W0[:D], v = f @ W0[D:]. Tiny GEMM.
// ---------------------------------------------------------------------------
__global__ void proj_kernel(const float* feat1, const float* feat2,
                            const float* w0, float* ws) {
  __shared__ float sf[Dd];
  int blk = blockIdx.x;
  int i = blk % Nn, bf = blk / Nn, f = bf & 1, b = bf >> 1;
  const float* fp = (f == 0 ? feat1 : feat2) + (b * Nn + i) * Dd;
  int c = threadIdx.x;
  sf[c] = fp[c];
  __syncthreads();
  float au = 0.f, av = 0.f;
  for (int d = 0; d < Dd; ++d) {
    float x = sf[d];
    au += x * w0[d * Dd + c];
    av += x * w0[(Dd + d) * Dd + c];
  }
  (ws + WS_U)[(bf * Nn + i) * Dd + c] = au;
  (ws + WS_V)[(bf * Nn + i) * Dd + c] = av;
}

// ---------------------------------------------------------------------------
// Per (b,frame,i): sim row -> top-3 (ties: earliest index) -> gather + absorb.
// b1 dropped: constant shift is rank-invariant for top_k.
// ---------------------------------------------------------------------------
__global__ void simtopk_kernel(const float* feat1, const float* feat2, float* ws) {
  __shared__ float su[Dd];
  __shared__ float ssim[Nn];
  __shared__ int   sidx[Kk];
  int blk = blockIdx.x;
  int i = blk % Nn, bf = blk / Nn, f = bf & 1, b = bf >> 1;
  const float* u     = ws + WS_U + (bf * Nn + i) * Dd;
  const float* vbase = ws + WS_V + bf * Nn * Dd;
  const float* muls = ws;       const float* adds = ws + 128;
  const float* w1s  = ws + 256;
  int tid = threadIdx.x;
  su[tid] = u[tid];
  __syncthreads();
  for (int jj = 0; jj < 2; ++jj) {
    int j = tid + jj * Dd;
    const float* vr = vbase + j * Dd;
    float dot = 0.f;
    for (int c = 0; c < Dd; ++c) {
      float t = (su[c] + vr[c]) * muls[c] + adds[c];
      t = fmaxf(t, 0.f);
      dot += t * w1s[c];
    }
    ssim[j] = (j == i) ? -INFINITY : dot;
  }
  __syncthreads();
  if (tid == 0) {
    float v0 = -INFINITY, v1 = -INFINITY, v2 = -INFINITY;
    int x0 = 0, x1 = 0, x2 = 0;
    for (int j = 0; j < Nn; ++j) {
      float s = ssim[j];
      if (s > v0)      { v2 = v1; x2 = x1; v1 = v0; x1 = x0; v0 = s; x0 = j; }
      else if (s > v1) { v2 = v1; x2 = x1; v1 = s;  x1 = j; }
      else if (s > v2) { v2 = s;  x2 = j; }
    }
    sidx[0] = x0; sidx[1] = x1; sidx[2] = x2;
  }
  __syncthreads();
  const float* fbase = (f == 0 ? feat1 : feat2) + b * Nn * Dd;
  float f0 = fbase[i * Dd + tid];
  float nsum = 0.f;
  float* nbr = ws + WS_NBR;
  for (int k = 0; k < Kk; ++k) {
    float nv = fbase[sidx[k] * Dd + tid];
    nbr[((bf * Nn + i) * Kk + k) * Dd + tid] = nv;
    nsum += nv;
  }
  (ws + WS_ABS)[(bf * Nn + i) * Dd + tid] = f0 + nsum * (1.f / 6.f);  // +0.5*mean
}

// ---------------------------------------------------------------------------
// Heavy kernel: [524288 x 128] |x-y| rows through shared 128x128 Linear via
// v_wmma_f32_16x16x32_f16, fused BN+ReLU+dot(w1)+sigmoid epilogue.
// Block = 16x16 (i,j) tile of one batch; 8 waves; each wave: 2 i's x 16 j's
// x 4 variants = 8 row-tiles of 16 rows = 256 WMMA.
// W0's 32 B-fragments are hoisted into 256 VGPRs once -> the hot loop is
// back-to-back WMMA with no LDS waits.
// ---------------------------------------------------------------------------
__global__ __launch_bounds__(256) void cls_kernel(const float* __restrict__ w0g,
                                                  const float* __restrict__ b1g,
                                                  const float* __restrict__ ws,
                                                  float* __restrict__ out) {
  // B fragments of W0, swizzled so each lane's 16 halves are contiguous:
  // [frag = nt*4+kk][lane][e]  (32 frags * 32 lanes * 16 halves = 32 KB)
  __shared__ __attribute__((aligned(32))) _Float16 sW0[32 * 32 * 16];
  // Wave-private A staging: 16 rows x 128 cols f16 per wave (32 KB)
  __shared__ __attribute__((aligned(32))) _Float16 stag[8][16][Dd];

  int tid = threadIdx.x;
  int blk = blockIdx.x;
  int b = blk >> 8, rem = blk & 255;
  int i0 = (rem >> 4) << 4, j0 = (rem & 15) << 4;

  // Pre-swizzle W0 (k-major [128][128] in global) into WMMA B fragments.
  // B layout (32x16, f16): lanes 0-15 col N=lane, halves K = {0..7,16..23};
  // lanes 16-31 col N=lane-16, halves K = {8..15,24..31} (+32*kk).
  for (int p = 0; p < 4; ++p) {
    int fl = tid * 4 + p;
    int lane_ = fl & 31, frag = fl >> 5;
    int nt = frag >> 2, kk = frag & 3;
    int base = (lane_ < 16) ? 0 : 8;
    int n = nt * 16 + (lane_ & 15);
    for (int e = 0; e < 16; ++e) {
      int kloc = (e < 8) ? (base + e) : (16 + base + (e - 8));
      sW0[fl * 16 + e] = (_Float16)w0g[(kk * 32 + kloc) * Dd + n];
    }
  }
  __syncthreads();

  int wave = tid >> 5, lane = tid & 31;

  // Hoist all 32 B fragments into registers (invariant over the t loop).
  v16h bfr[32];
#pragma unroll
  for (int fg = 0; fg < 32; ++fg)
    bfr[fg] = *(const v16h*)(&sW0[(fg * 32 + lane) * 16]);

  const float* mulc = ws + 384; const float* addc = ws + 512;
  const float* w1c  = ws + 640;
  float mul_r[8], add_r[8], w1_r[8];
#pragma unroll
  for (int nt = 0; nt < 8; ++nt) {
    int col = nt * 16 + (lane & 15);          // epilogue column of this lane
    mul_r[nt] = mulc[col]; add_r[nt] = addc[col]; w1_r[nt] = w1c[col];
  }
  float b1 = b1g[0];

  const float* a1 = ws + WS_ABS + ((b * 2 + 0) * Nn) * Dd;
  const float* a2 = ws + WS_ABS + ((b * 2 + 1) * Nn) * Dd;
  const float* n1 = ws + WS_NBR + ((b * 2 + 0) * Nn) * Kk * Dd;
  const float* n2 = ws + WS_NBR + ((b * 2 + 1) * Nn) * Kk * Dd;

  for (int t = 0; t < 8; ++t) {
    // ---- stage 16 rows of |x-y| as f16 (row = 4*pair + variant) ----
    {
      int row = lane >> 1, half = lane & 1, col0 = half * 64;
      int q = 4 * t + (row >> 2), v = row & 3;
      int di = q >> 4, jj = q & 15;
      int ii = i0 + 2 * wave + di, jcol = j0 + jj;
      const float* px = (v == 0) ? (a1 + ii * Dd)
                                 : (n1 + (ii * Kk + (v - 1)) * Dd);
      const float* py = (v == 0) ? (a2 + jcol * Dd)
                                 : (n2 + (jcol * Kk + (v - 1)) * Dd);
      for (int c = 0; c < 64; c += 4) {
        float4 xa = *(const float4*)(px + col0 + c);
        float4 ya = *(const float4*)(py + col0 + c);
        stag[wave][row][col0 + c + 0] = (_Float16)fabsf(xa.x - ya.x);
        stag[wave][row][col0 + c + 1] = (_Float16)fabsf(xa.y - ya.y);
        stag[wave][row][col0 + c + 2] = (_Float16)fabsf(xa.z - ya.z);
        stag[wave][row][col0 + c + 3] = (_Float16)fabsf(xa.w - ya.w);
      }
    }
    __syncthreads();  // barrier + compiler fence for cross-lane LDS RAW

    // ---- load A fragments (16x32 f16 per K-step): two b128 runs/lane ----
    // lanes 0-15: row M=lane, K halves {0..7,16..23}; lanes 16-31: M=lane-16,
    // K halves {8..15,24..31} (+32*kk).
    v16h afr[4];
    {
      int M = lane & 15, base = (lane >> 4) * 8;
#pragma unroll
      for (int kk = 0; kk < 4; ++kk) {
        union { v16h w; v8h h[2]; } u;
        u.h[0] = *(const v8h*)(&stag[wave][M][kk * 32 + base]);
        u.h[1] = *(const v8h*)(&stag[wave][M][kk * 32 + base + 16]);
        afr[kk] = u.w;
      }
    }

    // ---- GEMM: 4 K-steps x 8 N-tiles of back-to-back WMMA ----
    v8f acc[8];
#pragma unroll
    for (int nt = 0; nt < 8; ++nt) acc[nt] = (v8f){};
#pragma unroll
    for (int kk = 0; kk < 4; ++kk)
#pragma unroll
      for (int nt = 0; nt < 8; ++nt)
        acc[nt] = __builtin_amdgcn_wmma_f32_16x16x32_f16(
            false, afr[kk], false, bfr[nt * 4 + kk], (short)0, acc[nt],
            false, false);

    // ---- fused epilogue: BN+ReLU+dot(w1), per-lane partial row sums ----
    // C layout: VGPR r -> row M = r + 8*(lane/16), col N = nt*16 + lane%16
    float partial[8];
#pragma unroll
    for (int r = 0; r < 8; ++r) partial[r] = 0.f;
#pragma unroll
    for (int nt = 0; nt < 8; ++nt)
#pragma unroll
      for (int r = 0; r < 8; ++r) {
        float h = acc[nt][r] * mul_r[nt] + add_r[nt];  // bias folded into add
        h = fmaxf(h, 0.f);
        partial[r] += h * w1_r[nt];
      }

    // ---- reduce per-row partials across the 16 lanes sharing those rows ----
#pragma unroll
    for (int off = 8; off > 0; off >>= 1)
#pragma unroll
      for (int r = 0; r < 8; ++r)
        partial[r] += __shfl_xor(partial[r], off, 16);

    // lanes 0 / 16 hold rows 0-7 / 8-15 => 2 pairs (x4 variants) each
    if ((lane & 15) == 0) {
      int pbase = 4 * t + ((lane >> 4) ? 2 : 0);
      for (int pp = 0; pp < 2; ++pp) {
        float sg[4];
        for (int v = 0; v < 4; ++v) {
          float lg = partial[pp * 4 + v] + b1;
          sg[v] = 1.f / (1.f + __expf(-lg));
        }
        int q = pbase + pp;
        int ii = i0 + 2 * wave + (q >> 4);
        int jcol = j0 + (q & 15);
        // 0.5*(sig_anchor + mean_k sig_k)
        out[(b * Nn + ii) * Nn + jcol] =
            0.5f * sg[0] + (sg[1] + sg[2] + sg[3]) * (1.f / 6.f);
      }
    }
  }
}

// ---------------------------------------------------------------------------
extern "C" void kernel_launch(void* const* d_in, const int* in_sizes, int n_in,
                              void* d_out, int out_size, void* d_ws, size_t ws_size,
                              hipStream_t stream) {
  (void)in_sizes; (void)n_in; (void)out_size; (void)ws_size;
  const float* feat1  = (const float*)d_in[0];
  const float* feat2  = (const float*)d_in[1];
  const float* sim_w0 = (const float*)d_in[2];
  const float* sim_b0 = (const float*)d_in[3];
  const float* sim_g  = (const float*)d_in[4];
  const float* sim_be = (const float*)d_in[5];
  const float* sim_m  = (const float*)d_in[6];
  const float* sim_v  = (const float*)d_in[7];
  const float* sim_w1 = (const float*)d_in[8];
  // d_in[9] = sim_b1: constant shift, rank-invariant for top_k -> unused
  const float* cls_w0 = (const float*)d_in[10];
  const float* cls_b0 = (const float*)d_in[11];
  const float* cls_g  = (const float*)d_in[12];
  const float* cls_be = (const float*)d_in[13];
  const float* cls_m  = (const float*)d_in[14];
  const float* cls_v  = (const float*)d_in[15];
  const float* cls_w1 = (const float*)d_in[16];
  const float* cls_b1 = (const float*)d_in[17];
  float* ws  = (float*)d_ws;
  float* out = (float*)d_out;

  fold_kernel<<<1, 128, 0, stream>>>(sim_g, sim_be, sim_m, sim_v, sim_b0, sim_w1,
                                     cls_g, cls_be, cls_m, cls_v, cls_b0, cls_w1, ws);
  proj_kernel<<<BSz * 2 * Nn, 128, 0, stream>>>(feat1, feat2, sim_w0, ws);
  simtopk_kernel<<<BSz * 2 * Nn, 128, 0, stream>>>(feat1, feat2, ws);
  cls_kernel<<<BSz * 16 * 16, 256, 0, stream>>>(cls_w0, cls_b1, ws, out);
}
